// SCNLSTM_76287209112239
// MI455X (gfx1250) — matrix operand order repaired and checked by reading, not compiled, not run on hardware
//
#include <hip/hip_runtime.h>

// SCN-LSTM for MI455X (gfx1250): bf16 WMMA GEMMs, swizzled LDS A-tiles,
// async global->LDS double buffering. B=64, T=1024, NX=NH=512, NK=300, NL=2.

#define NB   64
#define NH_  512
#define TT   1024
#define NKEY 300
#define KDIM 512

typedef __attribute__((ext_vector_type(16))) __bf16 v16bf;
typedef __attribute__((ext_vector_type(8)))  float  v8f;

#define AS1 __attribute__((address_space(1)))
#define AS3 __attribute__((address_space(3)))

// exact pointee type expected by the async-LDS builtins (per hipcc diagnostic)
typedef int v4i __attribute__((__vector_size__(16)));
typedef AS1 v4i* gv4p;   // global (prints as "__device__ int4 *")
typedef AS3 v4i* lv4p;   // LDS    (prints as "__shared__ int4 *")

#if defined(__gfx1250__) && \
    __has_builtin(__builtin_amdgcn_global_load_async_to_lds_b128) && \
    __has_builtin(__builtin_amdgcn_s_wait_asynccnt)
#define USE_ASYNC_LDS 1
#else
#define USE_ASYNC_LDS 0
#endif

__device__ __forceinline__ unsigned short f2bf(float f) {
  return __builtin_bit_cast(unsigned short, (__bf16)f);   // v_cvt, RTNE
}
__device__ __forceinline__ unsigned pack2bf(float a, float b) {
  return (unsigned)f2bf(a) | ((unsigned)f2bf(b) << 16);
}
__device__ __forceinline__ float sigm(float x) { return 1.0f / (1.0f + __expf(-x)); }

// ---------------------------------------------------------------------------
// One-time: W[4][K][512] f32 -> Wt[4][512][K] bf16 (K-contiguous per column)
__global__ void wtrans_kernel(const float* __restrict__ W,
                              unsigned short* __restrict__ Wt) {
  long idx = (long)blockIdx.x * blockDim.x + threadIdx.x;
  if (idx >= 4L * KDIM * NH_) return;
  int n = (int)(idx % NH_);
  long t = idx / NH_;
  int k = (int)(t % KDIM);
  int g = (int)(t / KDIM);
  Wt[((long)g * NH_ + n) * KDIM + k] = f2bf(W[idx]);
}

// One-time: tb/ub = einsum('bk,gkf->gbf', tag, Wb)  (K=300, tiny)
__global__ void tagproj_kernel(const float* __restrict__ tag,
                               const float* __restrict__ Wb,
                               float* __restrict__ out) {
  int idx = blockIdx.x * blockDim.x + threadIdx.x;   // 4*64*512
  int n = idx & (NH_ - 1);
  int t = idx >> 9;
  int b = t & (NB - 1);
  int g = t >> 6;
  const float* tg = tag + b * NKEY;
  const float* w  = Wb + (long)g * NKEY * NH_ + n;
  float s = 0.f;
  for (int k = 0; k < NKEY; ++k) s += tg[k] * w[(long)k * NH_];
  out[idx] = s;
}

// One-time: state init (h0/c0 are [B][NL][NH]; internal state is [NL][B][NH])
__global__ void init_state_kernel(const float* __restrict__ h0,
                                  const float* __restrict__ c0,
                                  unsigned short* __restrict__ hb0,
                                  unsigned short* __restrict__ hb1,
                                  float* __restrict__ hf0, float* __restrict__ hf1,
                                  float* __restrict__ cf0, float* __restrict__ cf1) {
  int idx = blockIdx.x * blockDim.x + threadIdx.x;   // 64*512
  int n = idx & (NH_ - 1);
  int b = idx >> 9;
  float hl0 = h0[((long)b * 2 + 0) * NH_ + n];
  float hl1 = h0[((long)b * 2 + 1) * NH_ + n];
  hb0[idx] = f2bf(hl0); hf0[idx] = hl0;
  hb1[idx] = f2bf(hl1); hf1[idx] = hl1;
  cf0[idx] = c0[((long)b * 2 + 0) * NH_ + n];
  cf1[idx] = c0[((long)b * 2 + 1) * NH_ + n];
}

// ---------------------------------------------------------------------------
// C[g](64x512) = A(64x512) @ Bt[g]^T, optional scale, bf16 or f32(+=) output.
// grid = (4, 1, 4 gates), block = 256 (8 waves). WG tile: 64(M) x 128(N).
// LDS holds the 64x32 A chunk pre-swizzled into WMMA A-fragment order:
// slot (mt*32 + lane)*16 .. +15 = lane's 16 contiguous bf16 for mtile mt.
template <bool A_F32, bool SCALE_TB, bool OUT_BF, bool OUT_F32, bool ACCUM>
__global__ __launch_bounds__(256)
void gemm_kernel(const void* __restrict__ Abase, long a_gstride, int a_rstride,
                 const unsigned short* __restrict__ Bt,
                 const float* __restrict__ scale,
                 unsigned short* __restrict__ out_bf,
                 float* __restrict__ out_f32) {
  __shared__ unsigned short As[2][64 * 32];            // 2 x 4 KB (double buffer)

  const int tid   = threadIdx.x;
  const int lane  = tid & 31;
  const int wave  = tid >> 5;
  const int mtile = wave & 3;
  const int ngrp  = wave >> 2;
  const int g     = blockIdx.z;
  const int nt0   = blockIdx.x * 8 + ngrp * 4;         // first of 4 N-tiles
  const int half  = lane >> 4;

  const float*          a_f = (const float*)Abase;
  const unsigned short* a_h = (const unsigned short*)Abase + (long)g * a_gstride;

  // Cooperative A staging: thread -> (row, 8 contiguous K), swizzled LDS slot.
  const int lrow  = tid >> 2;
  const int lcol  = (tid & 3) * 8;
  const int s_mt  = lrow >> 4;
  const int s_m   = lrow & 15;
  const int s_hf  = lcol >> 4;
  const int s_eb  = lcol & 8;
  const int sslot = ((s_mt * 32 + s_hf * 16 + s_m) * 16) + s_eb;  // halfword idx

  v8f acc0, acc1, acc2, acc3;
#pragma unroll
  for (int r = 0; r < 8; ++r) { acc0[r] = 0.f; acc1[r] = 0.f; acc2[r] = 0.f; acc3[r] = 0.f; }

  // B fragment pointers: 16 contiguous bf16 per lane (32B), advance 32/step.
  const long bbase = ((long)g * NH_ + (lane & 15)) * KDIM + 16 * half;
  const unsigned short* bp0 = Bt + bbase + (long)(nt0 + 0) * 16 * KDIM;
  const unsigned short* bp1 = Bt + bbase + (long)(nt0 + 1) * 16 * KDIM;
  const unsigned short* bp2 = Bt + bbase + (long)(nt0 + 2) * 16 * KDIM;
  const unsigned short* bp3 = Bt + bbase + (long)(nt0 + 3) * 16 * KDIM;

#if USE_ASYNC_LDS
  if (!A_F32) {
    // prime chunk 0
    __builtin_amdgcn_global_load_async_to_lds_b128(
        (gv4p)(unsigned long long)(const void*)(a_h + (long)lrow * a_rstride + lcol),
        (lv4p)(unsigned)(unsigned long long)(const void*)&As[0][sslot], 0, 0);
  }
#endif

  for (int k0 = 0; k0 < KDIM; k0 += 32) {
    const int cur = (k0 >> 5) & 1;
    if (A_F32) {
      const float* src = a_f + (long)lrow * a_rstride + k0 + lcol;
      float4 f0 = *(const float4*)src;
      float4 f1 = *(const float4*)(src + 4);
      uint4 pk;
      pk.x = pack2bf(f0.x, f0.y);
      pk.y = pack2bf(f0.z, f0.w);
      pk.z = pack2bf(f1.x, f1.y);
      pk.w = pack2bf(f1.z, f1.w);
      *(uint4*)&As[0][sslot] = pk;
    } else {
#if USE_ASYNC_LDS
      if (k0 + 32 < KDIM) {   // issue next chunk into other buffer
        __builtin_amdgcn_global_load_async_to_lds_b128(
            (gv4p)(unsigned long long)(const void*)
                (a_h + (long)lrow * a_rstride + k0 + 32 + lcol),
            (lv4p)(unsigned)(unsigned long long)(const void*)&As[1 - cur][sslot],
            0, 0);
        __builtin_amdgcn_s_wait_asynccnt(1);   // cur chunk done (in-order)
      } else {
        __builtin_amdgcn_s_wait_asynccnt(0);
      }
#else
      *(uint4*)&As[cur][sslot] =
          *(const uint4*)(a_h + (long)lrow * a_rstride + k0 + lcol);
#endif
    }
    __syncthreads();

    const int buf = A_F32 ? 0 : cur;
    v16bf av = *(const v16bf*)&As[buf][(mtile * 32 + lane) * 16];
    v16bf b0 = *(const v16bf*)(bp0 + k0);
    v16bf b1 = *(const v16bf*)(bp1 + k0);
    v16bf b2 = *(const v16bf*)(bp2 + k0);
    v16bf b3 = *(const v16bf*)(bp3 + k0);
    if (k0 + 32 < KDIM) __builtin_prefetch(bp0 + k0 + 32, 0, 1);

    acc0 = __builtin_amdgcn_wmma_f32_16x16x32_bf16(false, av, false, b0, (short)0, acc0, false, false);
    acc1 = __builtin_amdgcn_wmma_f32_16x16x32_bf16(false, av, false, b1, (short)0, acc1, false, false);
    acc2 = __builtin_amdgcn_wmma_f32_16x16x32_bf16(false, av, false, b2, (short)0, acc2, false, false);
    acc3 = __builtin_amdgcn_wmma_f32_16x16x32_bf16(false, av, false, b3, (short)0, acc3, false, false);
    __syncthreads();
  }

  // Epilogue: C layout — VGPR r, lane: M = r + 8*half, N = lane&15
  v8f* accs[4] = {&acc0, &acc1, &acc2, &acc3};
#pragma unroll
  for (int j = 0; j < 4; ++j) {
    int n = (nt0 + j) * 16 + (lane & 15);
#pragma unroll
    for (int r = 0; r < 8; ++r) {
      int m = mtile * 16 + r + 8 * half;
      long idx = (((long)g * NB) + m) * NH_ + n;
      float v = (*accs[j])[r];
      if (SCALE_TB) v *= scale[idx];
      if (OUT_BF)   out_bf[idx] = f2bf(v);
      if (OUT_F32)  out_f32[idx] = ACCUM ? (out_f32[idx] + v) : v;
    }
  }
}

// ---------------------------------------------------------------------------
// Gate nonlinearity + cell/hidden update (reference: sigmoid on all 4 gates)
__global__ void pointwise_kernel(const float* __restrict__ gpre,
                                 const float* __restrict__ bias,
                                 float* __restrict__ c_state,
                                 unsigned short* __restrict__ h_bf,
                                 float* __restrict__ h_f32,
                                 float* __restrict__ seq_out, int t) {
  int idx = blockIdx.x * blockDim.x + threadIdx.x;   // 64*512
  int n = idx & (NH_ - 1);
  int b = idx >> 9;
  long base = (long)b * NH_ + n;
  float gi = sigm(gpre[base + 0L * NB * NH_] + bias[0 * NH_ + n]);
  float ga = sigm(gpre[base + 1L * NB * NH_] + bias[1 * NH_ + n]);
  float go = sigm(gpre[base + 2L * NB * NH_] + bias[2 * NH_ + n]);
  float gc = sigm(gpre[base + 3L * NB * NH_] + bias[3 * NH_ + n]);
  float c = gi * gc + ga * c_state[idx];
  float h = go * tanhf(c);
  c_state[idx] = c;
  h_bf[idx]    = f2bf(h);
  h_f32[idx]   = h;
  if (seq_out) seq_out[((long)b * TT + t) * NH_ + n] = h;
}

__global__ void finalize_kernel(const float* __restrict__ hf0,
                                const float* __restrict__ hf1,
                                const float* __restrict__ cf0,
                                const float* __restrict__ cf1,
                                float* __restrict__ out_h,
                                float* __restrict__ out_c) {
  int idx = blockIdx.x * blockDim.x + threadIdx.x;   // 64*512
  int n = idx & (NH_ - 1);
  int b = idx >> 9;
  out_h[((long)b * 2 + 0) * NH_ + n] = hf0[idx];
  out_h[((long)b * 2 + 1) * NH_ + n] = hf1[idx];
  out_c[((long)b * 2 + 0) * NH_ + n] = cf0[idx];
  out_c[((long)b * 2 + 1) * NH_ + n] = cf1[idx];
}

// ---------------------------------------------------------------------------
extern "C" void kernel_launch(void* const* d_in, const int* in_sizes, int n_in,
                              void* d_out, int out_size, void* d_ws, size_t ws_size,
                              hipStream_t stream) {
  (void)in_sizes; (void)n_in; (void)out_size; (void)ws_size;

  const float* inps = (const float*)d_in[0];
  const float* h0   = (const float*)d_in[1];
  const float* c0   = (const float*)d_in[2];
  const float* tag  = (const float*)d_in[3];
  const float* Wa[2]   = {(const float*)d_in[4],  (const float*)d_in[11]};
  const float* Wb[2]   = {(const float*)d_in[5],  (const float*)d_in[12]};
  const float* Wc[2]   = {(const float*)d_in[6],  (const float*)d_in[13]};
  const float* Ua[2]   = {(const float*)d_in[7],  (const float*)d_in[14]};
  const float* Ub[2]   = {(const float*)d_in[8],  (const float*)d_in[15]};
  const float* Uc[2]   = {(const float*)d_in[9],  (const float*)d_in[16]};
  const float* bias[2] = {(const float*)d_in[10], (const float*)d_in[17]};

  char* ws = (char*)d_ws;
  size_t off = 0;
  auto alloc = [&](size_t bytes) -> void* {
    off = (off + 255) & ~(size_t)255;
    void* p = ws + off;
    off += bytes;
    return p;
  };

  const size_t WBYTES = 4ull * NH_ * KDIM * sizeof(unsigned short);  // 2 MB
  unsigned short* Wct[2]; unsigned short* Wat[2];
  unsigned short* Uct[2]; unsigned short* Uat[2];
  for (int l = 0; l < 2; ++l) {
    Wct[l] = (unsigned short*)alloc(WBYTES);
    Wat[l] = (unsigned short*)alloc(WBYTES);
    Uct[l] = (unsigned short*)alloc(WBYTES);
    Uat[l] = (unsigned short*)alloc(WBYTES);
  }
  const size_t GB4 = 4ull * NB * NH_;
  float* tb[2]; float* ub[2];
  for (int l = 0; l < 2; ++l) {
    tb[l] = (float*)alloc(GB4 * sizeof(float));
    ub[l] = (float*)alloc(GB4 * sizeof(float));
  }
  unsigned short* Mx   = (unsigned short*)alloc(GB4 * sizeof(unsigned short));
  float*          gpre = (float*)alloc(GB4 * sizeof(float));
  const size_t BH = (size_t)NB * NH_;
  unsigned short* hbf[2] = {(unsigned short*)alloc(BH * 2), (unsigned short*)alloc(BH * 2)};
  float* hf[2] = {(float*)alloc(BH * 4), (float*)alloc(BH * 4)};
  float* cf[2] = {(float*)alloc(BH * 4), (float*)alloc(BH * 4)};

  // --- setup ---
  dim3 blk(256);
  dim3 wtg((unsigned)((4L * NH_ * KDIM + 255) / 256));
  for (int l = 0; l < 2; ++l) {
    wtrans_kernel<<<wtg, blk, 0, stream>>>(Wc[l], Wct[l]);
    wtrans_kernel<<<wtg, blk, 0, stream>>>(Wa[l], Wat[l]);
    wtrans_kernel<<<wtg, blk, 0, stream>>>(Uc[l], Uct[l]);
    wtrans_kernel<<<wtg, blk, 0, stream>>>(Ua[l], Uat[l]);
  }
  dim3 tpg((unsigned)(GB4 / 256));
  for (int l = 0; l < 2; ++l) {
    tagproj_kernel<<<tpg, blk, 0, stream>>>(tag, Wb[l], tb[l]);
    tagproj_kernel<<<tpg, blk, 0, stream>>>(tag, Ub[l], ub[l]);
  }
  dim3 pwg((unsigned)(BH / 256));
  init_state_kernel<<<pwg, blk, 0, stream>>>(h0, c0, hbf[0], hbf[1],
                                             hf[0], hf[1], cf[0], cf[1]);

  float* out_seq = (float*)d_out;
  float* out_h   = out_seq + (long)NB * TT * NH_;
  float* out_c   = out_h + (long)NB * 2 * NH_;

  dim3 gg(4, 1, 4);                 // N/128 tiles x 4 gates
  const long MGS = (long)NB * NH_;  // Mx per-gate stride

  for (int t = 0; t < TT; ++t) {
    // ---- layer 0 ----
    gemm_kernel<true,  true,  true,  false, false><<<gg, blk, 0, stream>>>(
        inps + (long)t * NH_, 0, TT * NH_, Wct[0], tb[0], Mx, nullptr);
    gemm_kernel<false, false, false, true,  false><<<gg, blk, 0, stream>>>(
        Mx, MGS, NH_, Wat[0], nullptr, nullptr, gpre);
    gemm_kernel<false, true,  true,  false, false><<<gg, blk, 0, stream>>>(
        hbf[0], 0, NH_, Uct[0], ub[0], Mx, nullptr);
    gemm_kernel<false, false, false, true,  true ><<<gg, blk, 0, stream>>>(
        Mx, MGS, NH_, Uat[0], nullptr, nullptr, gpre);
    pointwise_kernel<<<pwg, blk, 0, stream>>>(gpre, bias[0], cf[0], hbf[0],
                                              hf[0], nullptr, t);
    // ---- layer 1 (x = layer-0 h) ----
    gemm_kernel<false, true,  true,  false, false><<<gg, blk, 0, stream>>>(
        hbf[0], 0, NH_, Wct[1], tb[1], Mx, nullptr);
    gemm_kernel<false, false, false, true,  false><<<gg, blk, 0, stream>>>(
        Mx, MGS, NH_, Wat[1], nullptr, nullptr, gpre);
    gemm_kernel<false, true,  true,  false, false><<<gg, blk, 0, stream>>>(
        hbf[1], 0, NH_, Uct[1], ub[1], Mx, nullptr);
    gemm_kernel<false, false, false, true,  true ><<<gg, blk, 0, stream>>>(
        Mx, MGS, NH_, Uat[1], nullptr, nullptr, gpre);
    pointwise_kernel<<<pwg, blk, 0, stream>>>(gpre, bias[1], cf[1], hbf[1],
                                              hf[1], out_seq, t);
  }
  finalize_kernel<<<pwg, blk, 0, stream>>>(hf[0], hf[1], cf[0], cf[1],
                                           out_h, out_c);
}
